// MambaBlock_20624432955860
// MI455X (gfx1250) — matrix operand order repaired and checked by reading, not compiled, not run on hardware
//
#include <hip/hip_runtime.h>
#include <hip/hip_bf16.h>
#include <cstdint>
#include <cstddef>

// ---------------------------------------------------------------------------
// Types for gfx1250 WMMA
// ---------------------------------------------------------------------------
typedef __bf16 bf16_t;
typedef __attribute__((ext_vector_type(16))) __bf16 v16bf;
typedef __attribute__((ext_vector_type(8)))  __bf16 v8bf;
typedef __attribute__((ext_vector_type(8)))  float  v8f;

// Problem dimensions (fixed by the reference's setup_inputs)
static constexpr int BB    = 2;
static constexpr int LLEN  = 1024;
static constexpr int T     = BB * LLEN;          // 2048 tokens
static constexpr int DM    = 1024;               // d_model
static constexpr int DI    = 2048;               // d_inner
static constexpr int DXZ   = 2 * DI;             // 4096
static constexpr int NST   = 16;                 // SSM state size
static constexpr int DTR   = 64;                 // dt_rank
static constexpr int DPROJ = DTR + 2 * NST;      // 96
static constexpr int DFF   = 4096;

__device__ __forceinline__ bf16_t f2bf(float f) {
  unsigned u = __builtin_bit_cast(unsigned, f);
  u += 0x7fffu + ((u >> 16) & 1u);               // round-to-nearest-even
  unsigned short h = (unsigned short)(u >> 16);
  return __builtin_bit_cast(bf16_t, h);
}

__device__ __forceinline__ float sigmoid_f(float x) { return 1.f / (1.f + __expf(-x)); }
__device__ __forceinline__ float gelu_f(float x) {
  float x3 = x * x * x;
  return 0.5f * x * (1.f + tanhf(0.7978845608028654f * (x + 0.044715f * x3)));
}
__device__ __forceinline__ float softplus_f(float x) {
  return (x > 20.f) ? x : log1pf(__expf(x));
}

// ---------------------------------------------------------------------------
// Fragment loaders (16-bit WMMA operand layouts, CDNA5 ISA 7.12.2)
// A 16x32 (MxK): lane&15 = M row; lane>>4 selects K halves:
//   halfs[0..7]  = K = k0 + 8*(lane>>4) + 0..7
//   halfs[8..15] = K = k0 + 16 + 8*(lane>>4) + 0..7
// B 32x16 (KxN): lane&15 = N col; halfs use the same K order as A
// ---------------------------------------------------------------------------
__device__ __forceinline__ v16bf load_a_frag(const bf16_t* __restrict__ p) {
  union { v16bf v; v8bf h[2]; } u;
  u.h[0] = *(const v8bf*)(p);
  u.h[1] = *(const v8bf*)(p + 16);
  return u.v;
}
__device__ __forceinline__ v16bf load_b_frag(const bf16_t* __restrict__ p, int ldb) {
  v16bf v;
  #pragma unroll
  for (int s = 0; s < 8; ++s) {
    v[s] = p[(size_t)s * ldb];
    v[s + 8] = p[(size_t)(16 + s) * ldb];
  }
  return v;
}

// ---------------------------------------------------------------------------
// bf16 WMMA GEMM:  C[MxN] = act(A[MxK] @ W[KxN] + bias) + res
// A row-major (lda), W row-major KxN (ldb = N of weight), outputs dense MxN.
// NT = N tiles per wave (2 -> 32x32 wave tile, 4 -> 32x64 wave tile).
// ACT: 0 = none, 1 = gelu(tanh), 2 = softplus
// Block = 256 threads (8 waves, 2 M x 4 N), block tile 64 x (64*NT).
// Requires: M%64==0, N%32==0, K%32==0 (true for all uses here).
// ---------------------------------------------------------------------------
template<int NT, int ACT, bool HAS_BIAS, bool HAS_RES, bool OUT_F32, bool OUT_BF>
__global__ __launch_bounds__(256)
void gemm_bf16_wmma(const bf16_t* __restrict__ A, int lda,
                    const bf16_t* __restrict__ W, int ldb,
                    const float* __restrict__ bias,
                    const float* __restrict__ res,
                    float* __restrict__ Cf, bf16_t* __restrict__ Cb,
                    int M, int N, int K)
{
  const int lane = threadIdx.x & 31;
  const int wave = threadIdx.x >> 5;
  const int m0 = blockIdx.x * 64 + (wave & 1) * 32;
  const int n0 = blockIdx.y * (64 * NT) + (wave >> 1) * (16 * NT);
  if (m0 + 32 > M || n0 + 16 * NT > N) return;

  const int r   = lane & 15;
  const int hid = lane >> 4;

  const bf16_t* a0p = A + (size_t)(m0 + r) * lda + hid * 8;
  const bf16_t* a1p = a0p + (size_t)16 * lda;
  const bf16_t* bp  = W + (size_t)(hid * 8) * ldb + n0 + r;   // N col = n0 + r

  v8f acc[2][NT] = {};

  for (int k0 = 0; k0 < K; k0 += 32) {
    // Speculative prefetch of the next K slab (invalid addresses drop silently)
    __builtin_prefetch(a0p + k0 + 32, 0, 3);
    __builtin_prefetch(a1p + k0 + 32, 0, 3);
    __builtin_prefetch(bp + (size_t)(k0 + 32) * ldb, 0, 3);

    v16bf a0 = load_a_frag(a0p + k0);
    v16bf a1 = load_a_frag(a1p + k0);
    v16bf bfrag[NT];
    #pragma unroll
    for (int j = 0; j < NT; ++j)
      bfrag[j] = load_b_frag(bp + (size_t)k0 * ldb + 16 * j, ldb);
    #pragma unroll
    for (int j = 0; j < NT; ++j) {
      acc[0][j] = __builtin_amdgcn_wmma_f32_16x16x32_bf16(false, a0, false, bfrag[j], (short)0, acc[0][j], false, false);
      acc[1][j] = __builtin_amdgcn_wmma_f32_16x16x32_bf16(false, a1, false, bfrag[j], (short)0, acc[1][j], false, false);
    }
  }

  // C/D layout: VGPR g, lanes 0-15: (M = g,   N = lane)
  //                     lanes 16-31: (M = g+8, N = lane-16)
  #pragma unroll
  for (int ti = 0; ti < 2; ++ti) {
    #pragma unroll
    for (int tj = 0; tj < NT; ++tj) {
      const int mb  = m0 + 16 * ti + 8 * hid;
      const int col = n0 + 16 * tj + r;
      float bc = 0.f;
      if constexpr (HAS_BIAS) bc = bias[col];
      #pragma unroll
      for (int g = 0; g < 8; ++g) {
        float v = acc[ti][tj][g] + bc;
        if constexpr (ACT == 1) v = gelu_f(v);
        else if constexpr (ACT == 2) v = softplus_f(v);
        const size_t idx = (size_t)(mb + g) * N + col;
        if constexpr (HAS_RES) v += res[idx];
        if constexpr (OUT_F32) Cf[idx] = v;
        if constexpr (OUT_BF)  Cb[idx] = f2bf(v);
      }
    }
  }
}

// ---------------------------------------------------------------------------
// fp32 -> bf16 elementwise (weight conversion; layout preserved K x N)
// ---------------------------------------------------------------------------
__global__ __launch_bounds__(256)
void f32_to_bf16(const float* __restrict__ in, bf16_t* __restrict__ out, int n)
{
  int i = blockIdx.x * 256 + threadIdx.x;
  if (i < n) out[i] = f2bf(in[i]);
}

// ---------------------------------------------------------------------------
// LayerNorm over last dim D (one block per token), bf16 output
// ---------------------------------------------------------------------------
__global__ __launch_bounds__(256)
void layernorm_bf16(const float* __restrict__ x, const float* __restrict__ g,
                    const float* __restrict__ b, bf16_t* __restrict__ out, int D)
{
  const int token = blockIdx.x;
  const float* xr = x + (size_t)token * D;
  float s = 0.f, ss = 0.f;
  for (int i = threadIdx.x; i < D; i += 256) { float v = xr[i]; s += v; ss += v * v; }
  #pragma unroll
  for (int o = 16; o > 0; o >>= 1) { s += __shfl_xor(s, o, 32); ss += __shfl_xor(ss, o, 32); }
  __shared__ float red[2][8];
  const int wave = threadIdx.x >> 5, lane = threadIdx.x & 31;
  if (lane == 0) { red[0][wave] = s; red[1][wave] = ss; }
  __syncthreads();
  if (threadIdx.x == 0) {
    float ts = 0.f, tss = 0.f;
    #pragma unroll
    for (int w = 0; w < 8; ++w) { ts += red[0][w]; tss += red[1][w]; }
    float mean = ts / (float)D;
    float var  = tss / (float)D - mean * mean;
    red[0][0] = mean;
    red[1][0] = rsqrtf(var + 1e-5f);
  }
  __syncthreads();
  const float mean = red[0][0], rstd = red[1][0];
  bf16_t* orow = out + (size_t)token * D;
  for (int i = threadIdx.x; i < D; i += 256)
    orow[i] = f2bf((xr[i] - mean) * rstd * g[i] + b[i]);
}

// ---------------------------------------------------------------------------
// Depthwise causal conv (d_conv=4) + SiLU; fully parallel over (token, ch)
// xz is T x 4096; channel d reads column d (xm half). Outputs fp32 + bf16.
// ---------------------------------------------------------------------------
__global__ __launch_bounds__(256)
void conv_silu(const float* __restrict__ xz, const float* __restrict__ w_conv,
               const float* __restrict__ b_conv,
               float* __restrict__ xm_f, bf16_t* __restrict__ xm_b)
{
  const int idx = blockIdx.x * 256 + threadIdx.x;          // over T*DI
  const int d   = idx & (DI - 1);
  const int tok = idx >> 11;                               // DI == 2048
  const int t   = tok & (LLEN - 1);
  const int bb  = tok >> 10;
  float acc = b_conv[d];
  #pragma unroll
  for (int j = 0; j < 4; ++j) {
    const int tt = t - 3 + j;
    if (tt >= 0)
      acc += w_conv[d * 4 + j] * xz[((size_t)(bb * LLEN + tt)) * DXZ + d];
  }
  const float s = acc * sigmoid_f(acc);
  xm_f[(size_t)tok * DI + d] = s;
  xm_b[(size_t)tok * DI + d] = f2bf(s);
}

// ---------------------------------------------------------------------------
// Selective scan, one thread per (batch, channel). Sequential over L.
// Fused:  y = (sum_n h*C + u*Dp) * silu(z)   -> bf16 for the w_out GEMM.
// ---------------------------------------------------------------------------
__global__ __launch_bounds__(64)
void scan_kernel(const float* __restrict__ delta, const float* __restrict__ xm,
                 const float* __restrict__ xz, const float* __restrict__ proj,
                 const float* __restrict__ A_log, const float* __restrict__ Dp,
                 bf16_t* __restrict__ yg)
{
  const int bb = blockIdx.y;
  const int d  = blockIdx.x * 64 + threadIdx.x;            // < DI
  float a[NST], h[NST];
  #pragma unroll
  for (int n = 0; n < NST; ++n) { a[n] = -__expf(A_log[d * NST + n]); h[n] = 0.f; }
  const float dp = Dp[d];
  for (int t = 0; t < LLEN; ++t) {
    const size_t tok = (size_t)bb * LLEN + t;
    const float dl = delta[tok * DI + d];
    const float u  = xm[tok * DI + d];
    const float zz = xz[tok * DXZ + DI + d];
    const float* pr = proj + tok * DPROJ;
    const float du = dl * u;
    float y = 0.f;
    #pragma unroll
    for (int n = 0; n < NST; ++n) {
      const float Bc = pr[DTR + n];
      const float Cc = pr[DTR + NST + n];
      h[n] = __expf(dl * a[n]) * h[n] + du * Bc;
      y += h[n] * Cc;
    }
    y = (y + u * dp) * (zz * sigmoid_f(zz));
    yg[tok * DI + d] = f2bf(y);
  }
}

// ---------------------------------------------------------------------------
// Host-side orchestration
// ---------------------------------------------------------------------------
extern "C" void kernel_launch(void* const* d_in, const int* in_sizes, int n_in,
                              void* d_out, int out_size, void* d_ws, size_t ws_size,
                              hipStream_t stream)
{
  (void)in_sizes; (void)n_in; (void)out_size; (void)ws_size;

  const float* x      = (const float*)d_in[0];
  const float* ln1_g  = (const float*)d_in[1];
  const float* ln1_b  = (const float*)d_in[2];
  const float* w_in   = (const float*)d_in[3];
  const float* w_conv = (const float*)d_in[4];
  const float* b_conv = (const float*)d_in[5];
  const float* w_xproj= (const float*)d_in[6];
  const float* w_dt   = (const float*)d_in[7];
  const float* b_dt   = (const float*)d_in[8];
  const float* A_log  = (const float*)d_in[9];
  const float* Dp     = (const float*)d_in[10];
  const float* w_out  = (const float*)d_in[11];
  const float* ln2_g  = (const float*)d_in[12];
  const float* ln2_b  = (const float*)d_in[13];
  const float* w_ff1  = (const float*)d_in[14];
  const float* b_ff1  = (const float*)d_in[15];
  const float* w_ff2  = (const float*)d_in[16];
  const float* b_ff2  = (const float*)d_in[17];
  float* out = (float*)d_out;

  char* ws = (char*)d_ws;
  size_t off = 0;
  auto alloc = [&](size_t bytes) -> char* {
    char* p = ws + off;
    off += (bytes + 255) & ~(size_t)255;
    return p;
  };

  // Weights (bf16, row-major K x N, live for the whole call)
  bf16_t* wb_in    = (bf16_t*)alloc((size_t)DM * DXZ   * 2);
  bf16_t* wb_xproj = (bf16_t*)alloc((size_t)DI * DPROJ * 2);
  bf16_t* wb_dt    = (bf16_t*)alloc((size_t)DTR * DI   * 2);
  bf16_t* wb_out   = (bf16_t*)alloc((size_t)DI * DM    * 2);
  bf16_t* wb_ff1   = (bf16_t*)alloc((size_t)DM * DFF   * 2);
  bf16_t* wb_ff2   = (bf16_t*)alloc((size_t)DFF * DM   * 2);

  // Activations (with lifetime-safe aliasing)
  bf16_t* xln_bf  = (bf16_t*)alloc((size_t)T * DM * 2);   // reused as hln_bf
  float*  xz      = (float*) alloc((size_t)T * DXZ * 4);  // start reused as out_x
  float*  xm_f    = (float*) alloc((size_t)T * DI * 4);
  bf16_t* xm_bf   = (bf16_t*)alloc((size_t)T * DI * 2);   // reused as yg_bf
  float*  proj    = (float*) alloc((size_t)T * DPROJ * 4);
  bf16_t* proj_bf = (bf16_t*)alloc((size_t)T * DPROJ * 2);
  float*  delta   = (float*) alloc((size_t)T * DI * 4);   // reused as h1_bf

  bf16_t* hln_bf = xln_bf;            // dead after GEMM1
  float*  out_x  = xz;                // xz fully consumed by scan before GEMM4
  bf16_t* yg_bf  = xm_bf;             // xm_bf dead after GEMM2
  bf16_t* h1_bf  = (bf16_t*)delta;    // delta dead after scan (16MB each)

  auto cdiv = [](int a, int b) { return (a + b - 1) / b; };

  // 1) Weight conversion (fp32 -> bf16, layout preserved)
  f32_to_bf16<<<cdiv(DM * DXZ, 256),   256, 0, stream>>>(w_in,    wb_in,    DM * DXZ);
  f32_to_bf16<<<cdiv(DI * DPROJ, 256), 256, 0, stream>>>(w_xproj, wb_xproj, DI * DPROJ);
  f32_to_bf16<<<cdiv(DTR * DI, 256),   256, 0, stream>>>(w_dt,    wb_dt,    DTR * DI);
  f32_to_bf16<<<cdiv(DI * DM, 256),    256, 0, stream>>>(w_out,   wb_out,   DI * DM);
  f32_to_bf16<<<cdiv(DM * DFF, 256),   256, 0, stream>>>(w_ff1,   wb_ff1,   DM * DFF);
  f32_to_bf16<<<cdiv(DFF * DM, 256),   256, 0, stream>>>(w_ff2,   wb_ff2,   DFF * DM);

  // 2) LN1
  layernorm_bf16<<<T, 256, 0, stream>>>(x, ln1_g, ln1_b, xln_bf, DM);

  // 3) xz = xln @ w_in                              (2048 x 4096, K=1024)
  gemm_bf16_wmma<4, 0, false, false, true, false><<<dim3(T / 64, cdiv(DXZ, 256)), 256, 0, stream>>>(
      xln_bf, DM, wb_in, DXZ, nullptr, nullptr, xz, nullptr, T, DXZ, DM);

  // 4) depthwise causal conv + SiLU -> xm (fp32 + bf16)
  conv_silu<<<cdiv(T * DI, 256), 256, 0, stream>>>(xz, w_conv, b_conv, xm_f, xm_bf);

  // 5) proj = xm @ w_xproj                          (2048 x 96, K=2048)
  gemm_bf16_wmma<2, 0, false, false, true, true><<<dim3(T / 64, cdiv(DPROJ, 128)), 256, 0, stream>>>(
      xm_bf, DI, wb_xproj, DPROJ, nullptr, nullptr, proj, proj_bf, T, DPROJ, DI);

  // 6) delta = softplus(dt @ w_dt + b_dt)           (2048 x 2048, K=64)
  //    dt = proj[:, 0:64]  -> A = proj_bf with lda = 96
  gemm_bf16_wmma<4, 2, true, false, true, false><<<dim3(T / 64, cdiv(DI, 256)), 256, 0, stream>>>(
      proj_bf, DPROJ, wb_dt, DI, b_dt, nullptr, delta, nullptr, T, DI, DTR);

  // 7) selective scan + Dp skip + silu(z) gating -> yg (bf16)
  scan_kernel<<<dim3(DI / 64, BB), 64, 0, stream>>>(delta, xm_f, xz, proj, A_log, Dp, yg_bf);

  // 8) out_x = x + yg @ w_out                       (2048 x 1024, K=2048)
  gemm_bf16_wmma<4, 0, false, true, true, false><<<dim3(T / 64, cdiv(DM, 256)), 256, 0, stream>>>(
      yg_bf, DI, wb_out, DM, nullptr, x, out_x, nullptr, T, DM, DI);

  // 9) LN2
  layernorm_bf16<<<T, 256, 0, stream>>>(out_x, ln2_g, ln2_b, hln_bf, DM);

  // 10) h1 = gelu(hln @ w_ff1 + b_ff1)              (2048 x 4096, K=1024)
  gemm_bf16_wmma<4, 1, true, false, false, true><<<dim3(T / 64, cdiv(DFF, 256)), 256, 0, stream>>>(
      hln_bf, DM, wb_ff1, DFF, b_ff1, nullptr, nullptr, h1_bf, T, DFF, DM);

  // 11) out = out_x + h1 @ w_ff2 + b_ff2            (2048 x 1024, K=4096)
  gemm_bf16_wmma<4, 0, true, true, true, false><<<dim3(T / 64, cdiv(DM, 256)), 256, 0, stream>>>(
      h1_bf, DFF, wb_ff2, DM, b_ff2, out_x, out, nullptr, T, DM, DFF);
}
